// ImageTransformer_49108656063229
// MI455X (gfx1250) — matrix operand order repaired and checked by reading, not compiled, not run on hardware
//
#include <hip/hip_runtime.h>

// ---------------------------------------------------------------------------
// Model constants
// ---------------------------------------------------------------------------
#define Bsz   128
#define Cch   2
#define IMG   32
#define PS    4
#define Dm    256
#define Hh    8
#define HD    32
#define Ll    3
#define Ff    2048
#define NP    64
#define INP   4

// GEMM tiling: block tile 64(M) x 128(N) x 32(K); 8 waves, each 2x2 WMMA tiles
#define TM 64
#define TN 128
#define TK 32

typedef __attribute__((ext_vector_type(16))) __bf16 v16bf;
typedef __attribute__((ext_vector_type(8)))  float  v8f;
typedef __attribute__((ext_vector_type(4)))  unsigned int v4u;
typedef __attribute__((ext_vector_type(8)))  int v8i;
typedef __attribute__((ext_vector_type(4)))  int v4i;

#ifndef __has_builtin
#define __has_builtin(x) 0
#endif
#if __has_builtin(__builtin_amdgcn_tensor_load_to_lds) && __has_builtin(__builtin_amdgcn_s_wait_tensorcnt)
#define HAVE_TDM 1
#else
#define HAVE_TDM 0
#endif

static __device__ __forceinline__ unsigned short f2bf(float f) {
    unsigned u = __float_as_uint(f);
    unsigned r = u + 0x7FFFu + ((u >> 16) & 1u);   // round-to-nearest-even
    return (unsigned short)(r >> 16);
}

#if HAVE_TDM
// Issue one 2D TDM tile load: global (row stride stride_elems fp32) -> LDS,
// tile_x contiguous elements per row, tile_y rows. D# per CDNA5 ISA ch.8.
static __device__ __forceinline__ void tdm_load_2d(unsigned lds_off, const float* gaddr,
                                                   unsigned tile_x, unsigned tile_y,
                                                   long stride_elems) {
    unsigned long long ga = (unsigned long long)(size_t)gaddr;
    unsigned long long s0 = (unsigned long long)stride_elems;
    v4u g0;
    g0[0] = 1u;                                            // count=1 (valid user D#)
    g0[1] = lds_off;                                       // lds_addr (bytes)
    g0[2] = (unsigned)(ga & 0xFFFFFFFFu);                  // global_addr[31:0]
    g0[3] = (unsigned)((ga >> 32) & 0x1FFFFFFu) | (2u << 30); // addr[56:32] | type=2
    v8i g1;
    g1[0] = (int)(2u << 16);                               // data_size=4B, wg_mask=0
    g1[1] = (int)((tile_x & 0xFFFFu) << 16);               // tensor_dim0[15:0]
    g1[2] = (int)((tile_x >> 16) | ((tile_y & 0xFFFFu) << 16)); // dim0 hi | dim1 lo
    g1[3] = (int)((tile_y >> 16) | (tile_x << 16));        // dim1 hi | tile_dim0
    g1[4] = (int)(tile_y);                                 // tile_dim1 (tile_dim2=0)
    g1[5] = (int)(s0 & 0xFFFFFFFFu);                       // tensor_dim0_stride lo
    g1[6] = (int)((s0 >> 32) & 0xFFFFu);                   // stride0 hi | stride1 lo=0
    g1[7] = 0;
    v4i g2 = {0, 0, 0, 0};
    v4i g3 = {0, 0, 0, 0};
#if __clang_major__ >= 23
    v8i g4 = {0, 0, 0, 0, 0, 0, 0, 0};
    __builtin_amdgcn_tensor_load_to_lds(g0, g1, g2, g3, g4, 0);
#else
    __builtin_amdgcn_tensor_load_to_lds(g0, g1, g2, g3, 0);
#endif
}
#endif

// ---------------------------------------------------------------------------
// Generic strided batched GEMM, bf16 WMMA, fp32 in/out.
//   D[g,m,n] = alpha * sum_k A[g,m,k]*B[...] + bias[n]   (optional ReLU)
// A: k contiguous, row stride am.  B: transB=1 -> [N,K]; transB=0 -> [K,N].
// ---------------------------------------------------------------------------
struct GemmDesc {
    const float* A; const float* B; const float* bias; float* D;
    long a0, a1, a2, am;
    long b0, b1, b2, bx;
    long d0, d1, d2, dm;
    int  M, N, K;
    int  G0, G1, G2;
    float alpha;
    int  relu;
    int  transB;
};

__global__ void __launch_bounds__(256) k_gemm_bf16(GemmDesc g) {
    // raw fp32 staging tiles (TDM destination) + bf16 fragment-ordered tiles
    __shared__ __align__(16) float rawA[TM * TK];            // [m][k]  8KB
    __shared__ __align__(16) float rawB[TN * TK];            // [n][k] or [k][n]  16KB
    __shared__ __align__(32) unsigned int lA[TM / 16][32][8];  // 4KB
    __shared__ __align__(32) unsigned int lB[TN / 16][32][8];  // 8KB

    int gz = blockIdx.z;
    int i2 = gz % g.G2;
    int tq = gz / g.G2;
    int i1 = tq % g.G1;
    int i0 = tq / g.G1;
    long aoff = (long)i0 * g.a0 + (long)i1 * g.a1 + (long)i2 * g.a2;
    long boff = (long)i0 * g.b0 + (long)i1 * g.b1 + (long)i2 * g.b2;
    long doff = (long)i0 * g.d0 + (long)i1 * g.d1 + (long)i2 * g.d2;

    int m0 = blockIdx.y * TM;
    int n0 = blockIdx.x * TN;

    int tid  = threadIdx.x;
    int wave = tid >> 5;
    int lane = tid & 31;
    int tr0 = (wave >> 2) * 2;     // wave tile rows tr0, tr0+1   (of 4)
    int tc0 = (wave & 3) * 2;      // wave tile cols tc0, tc0+1   (of 8)

    v8f acc[2][2];
    #pragma unroll
    for (int i = 0; i < 2; ++i)
        #pragma unroll
        for (int j = 0; j < 2; ++j)
            acc[i][j] = v8f{0.f, 0.f, 0.f, 0.f, 0.f, 0.f, 0.f, 0.f};

    const int tilesM_ok = (m0 + TM <= g.M);
    const int tilesN_ok = (n0 + TN <= g.N);

    for (int kt = 0; kt < g.K; kt += TK) {
        int full = tilesM_ok && tilesN_ok && (kt + TK <= g.K);
        __syncthreads();   // raw tiles free for reuse
        if (full) {
#if HAVE_TDM
            if (wave == 0) {
                tdm_load_2d((unsigned)(size_t)(void*)rawA,
                            g.A + aoff + (long)m0 * g.am + kt, TK, TM, g.am);
                if (g.transB)
                    tdm_load_2d((unsigned)(size_t)(void*)rawB,
                                g.B + boff + (long)n0 * g.bx + kt, TK, TN, g.bx);
                else
                    tdm_load_2d((unsigned)(size_t)(void*)rawB,
                                g.B + boff + (long)kt * g.bx + n0, TN, TK, g.bx);
                __builtin_amdgcn_s_wait_tensorcnt((short)0);
            }
#else
            // direct staging, unguarded
            #pragma unroll
            for (int it = 0; it < 2; ++it) {
                int idx = tid + it * 256;            // 0..511, 4 floats each
                int mi = idx >> 3, kq = (idx & 7) * 4;
                const float* src = g.A + aoff + (long)(m0 + mi) * g.am + (kt + kq);
                rawA[mi * TK + kq + 0] = src[0]; rawA[mi * TK + kq + 1] = src[1];
                rawA[mi * TK + kq + 2] = src[2]; rawA[mi * TK + kq + 3] = src[3];
            }
            #pragma unroll
            for (int it = 0; it < 4; ++it) {
                int idx = tid + it * 256;            // 0..1023, 4 floats each
                if (g.transB) {
                    int ni = idx >> 3, kq = (idx & 7) * 4;
                    const float* src = g.B + boff + (long)(n0 + ni) * g.bx + (kt + kq);
                    rawB[ni * TK + kq + 0] = src[0]; rawB[ni * TK + kq + 1] = src[1];
                    rawB[ni * TK + kq + 2] = src[2]; rawB[ni * TK + kq + 3] = src[3];
                } else {
                    int ki = idx >> 5, nq = (idx & 31) * 4;
                    const float* src = g.B + boff + (long)(kt + ki) * g.bx + (n0 + nq);
                    rawB[ki * TN + nq + 0] = src[0]; rawB[ki * TN + nq + 1] = src[1];
                    rawB[ki * TN + nq + 2] = src[2]; rawB[ki * TN + nq + 3] = src[3];
                }
            }
#endif
        } else {
            // clamped, branchless staging with select-to-zero
            #pragma unroll
            for (int it = 0; it < 4; ++it) {
                int idx = tid + it * 256;            // 0..1023 : A pairs
                int mi = idx >> 4, k = (idx & 15) * 2;
                int gm = m0 + mi, kk = kt + k;
                int cm = (gm < g.M) ? gm : (g.M - 1);
                int c0 = (kk < g.K) ? kk : (g.K - 1);
                int c1 = (kk + 1 < g.K) ? (kk + 1) : (g.K - 1);
                long base = aoff + (long)cm * g.am;
                float f0 = g.A[base + c0];
                float f1 = g.A[base + c1];
                f0 = (gm < g.M && kk < g.K) ? f0 : 0.f;
                f1 = (gm < g.M && kk + 1 < g.K) ? f1 : 0.f;
                rawA[mi * TK + k] = f0; rawA[mi * TK + k + 1] = f1;
            }
            #pragma unroll
            for (int it = 0; it < 8; ++it) {
                int idx = tid + it * 256;            // 0..2047 : B pairs
                int ni = idx >> 4, k = (idx & 15) * 2;
                int gn = n0 + ni, kk = kt + k;
                int cn = (gn < g.N) ? gn : (g.N - 1);
                int c0 = (kk < g.K) ? kk : (g.K - 1);
                int c1 = (kk + 1 < g.K) ? (kk + 1) : (g.K - 1);
                float f0, f1;
                if (g.transB) {
                    long base = boff + (long)cn * g.bx;
                    f0 = g.B[base + c0]; f1 = g.B[base + c1];
                } else {
                    f0 = g.B[boff + (long)c0 * g.bx + cn];
                    f1 = g.B[boff + (long)c1 * g.bx + cn];
                }
                f0 = (gn < g.N && kk < g.K) ? f0 : 0.f;
                f1 = (gn < g.N && kk + 1 < g.K) ? f1 : 0.f;
                if (g.transB) { rawB[ni * TK + k] = f0; rawB[ni * TK + k + 1] = f1; }
                else          { rawB[k * TN + ni] = f0; rawB[(k + 1) * TN + ni] = f1; }
            }
        }
        __syncthreads();   // raw ready
        // ---- pack raw fp32 -> bf16 fragments in WMMA VGPR order ----
        #pragma unroll
        for (int it = 0; it < 4; ++it) {
            int idx = tid + it * 256;                // 0..1023 : A pairs
            int mi = idx >> 4, k = (idx & 15) * 2;
            float f0 = rawA[mi * TK + k], f1 = rawA[mi * TK + k + 1];
            unsigned u = (unsigned)f2bf(f0) | ((unsigned)f2bf(f1) << 16);
            int half = (k & 15) >> 3;
            int v = ((k & 7) >> 1) + ((k & 16) ? 4 : 0);
            lA[mi >> 4][half * 16 + (mi & 15)][v] = u;
        }
        #pragma unroll
        for (int it = 0; it < 8; ++it) {
            int idx = tid + it * 256;                // 0..2047 : B pairs
            int ni = idx >> 4, k = (idx & 15) * 2;
            float f0, f1;
            if (g.transB) { f0 = rawB[ni * TK + k]; f1 = rawB[ni * TK + k + 1]; }
            else          { f0 = rawB[k * TN + ni]; f1 = rawB[(k + 1) * TN + ni]; }
            unsigned u = (unsigned)f2bf(f0) | ((unsigned)f2bf(f1) << 16);
            int lhalf = (k & 15) >> 3;
            int v = ((k & 7) >> 1) + ((k & 16) ? 4 : 0);
            lB[ni >> 4][lhalf * 16 + (ni & 15)][v] = u;
        }
        __syncthreads();   // fragments ready
        // ---- 4 WMMAs per wave per K-step; 32B contiguous LDS frag reads ----
        v16bf a0 = *(const v16bf*)&lA[tr0][lane][0];
        v16bf a1 = *(const v16bf*)&lA[tr0 + 1][lane][0];
        v16bf b0 = *(const v16bf*)&lB[tc0][lane][0];
        v16bf b1 = *(const v16bf*)&lB[tc0 + 1][lane][0];
        acc[0][0] = __builtin_amdgcn_wmma_f32_16x16x32_bf16(false, a0, false, b0, (short)0, acc[0][0], false, false);
        acc[0][1] = __builtin_amdgcn_wmma_f32_16x16x32_bf16(false, a0, false, b1, (short)0, acc[0][1], false, false);
        acc[1][0] = __builtin_amdgcn_wmma_f32_16x16x32_bf16(false, a1, false, b0, (short)0, acc[1][0], false, false);
        acc[1][1] = __builtin_amdgcn_wmma_f32_16x16x32_bf16(false, a1, false, b1, (short)0, acc[1][1], false, false);
    }

    // ---- epilogue: lane(0-15)->N ; VGPR r -> M row (lanes>=16: +8) ----
    #pragma unroll
    for (int i = 0; i < 2; ++i) {
        int mbase = m0 + (tr0 + i) * 16 + ((lane >= 16) ? 8 : 0);
        #pragma unroll
        for (int j = 0; j < 2; ++j) {
            int ncol = n0 + (tc0 + j) * 16 + (lane & 15);
            if (ncol < g.N) {
                float bval = g.bias ? g.bias[ncol] : 0.f;
                #pragma unroll
                for (int r = 0; r < 8; ++r) {
                    int mm = mbase + r;
                    if (mm < g.M) {
                        float xv = acc[i][j][r] * g.alpha + bval;
                        if (g.relu) xv = fmaxf(xv, 0.f);
                        g.D[doff + (long)mm * g.dm + ncol] = xv;
                    }
                }
            }
        }
    }
}

// ---------------------------------------------------------------------------
// Elementwise / reduction kernels
// ---------------------------------------------------------------------------
__global__ void k_im2col_outer(const float* __restrict__ x, float* __restrict__ A) {
    int idx = blockIdx.x * 256 + threadIdx.x;           // 8192*32
    if (idx >= 8192 * 32) return;
    int k = idx & 31; int row = idx >> 5;
    int b = row >> 6; int p = row & 63;
    int r = p >> 3, hc = p & 7;
    int c = k >> 4; int pp = (k >> 2) & 3; int q = k & 3;
    A[idx] = x[((b * Cch + c) * IMG + (r * 4 + pp)) * IMG + (hc * 4 + q)];
}

__global__ void k_im2col_inner(const float* __restrict__ x, float* __restrict__ A) {
    int idx = blockIdx.x * 256 + threadIdx.x;           // 32768*8
    if (idx >= 32768 * 8) return;
    int k = idx & 7; int row = idx >> 3;
    int inp = row & 3; int t = row >> 2;
    int b = t & 127; int np = t >> 7;
    int R = np >> 3, Cc = np & 7;
    int r = inp >> 1, h = inp & 1;
    int c = k >> 2, i = (k >> 1) & 1, j = k & 1;
    A[idx] = x[((b * Cch + c) * IMG + (R * 4 + r * 2 + i)) * IMG + (Cc * 4 + h * 2 + j)];
}

__global__ void k_add_pos(float* __restrict__ y, const float* __restrict__ pos,
                          int n, int PR, float sign) {
    int idx = blockIdx.x * 256 + threadIdx.x;
    if (idx >= n) return;
    int d = idx & 255; int row = idx >> 8;
    y[idx] += sign * pos[(row % PR) * Dm + d];
}

__global__ void __launch_bounds__(256) k_add_ln(float* __restrict__ y,
                                                const float* __restrict__ sub,
                                                const float* __restrict__ gam,
                                                const float* __restrict__ bet) {
    __shared__ float s[256];
    long row = blockIdx.x;
    int t = threadIdx.x;
    float v = y[row * Dm + t] + sub[row * Dm + t];
    s[t] = v; __syncthreads();
    for (int o = 128; o > 0; o >>= 1) { if (t < o) s[t] += s[t + o]; __syncthreads(); }
    float mean = s[0] * (1.f / 256.f); __syncthreads();
    float dv = v - mean;
    s[t] = dv * dv; __syncthreads();
    for (int o = 128; o > 0; o >>= 1) { if (t < o) s[t] += s[t + o]; __syncthreads(); }
    float var = s[0] * (1.f / 256.f);
    y[row * Dm + t] = dv * rsqrtf(var + 1e-5f) * gam[t] + bet[t];
}

__global__ void __launch_bounds__(128) k_softmax128(float* __restrict__ Sc, int M, int causal) {
    __shared__ float s[128];
    long row = blockIdx.x;
    int i = (int)(row % (long)M);
    int j = threadIdx.x;
    float v = Sc[row * 128 + j];
    if (causal && j > i) v = -3.0e38f;
    s[j] = v; __syncthreads();
    for (int o = 64; o > 0; o >>= 1) { if (j < o) s[j] = fmaxf(s[j], s[j + o]); __syncthreads(); }
    float mx = s[0]; __syncthreads();
    float e = __expf(v - mx);
    s[j] = e; __syncthreads();
    for (int o = 64; o > 0; o >>= 1) { if (j < o) s[j] += s[j + o]; __syncthreads(); }
    Sc[row * 128 + j] = e / s[0];
}

__global__ void k_inner_lin(const float* __restrict__ y, const float* __restrict__ w,
                            const float* __restrict__ lb, float* __restrict__ codes) {
    int idx = blockIdx.x * 256 + threadIdx.x;           // 8192*256
    if (idx >= 8192 * 256) return;
    int d = idx & 255; long row = idx >> 8;             // row = np*B + b
    float acc = lb[0];
    #pragma unroll
    for (int inp = 0; inp < INP; ++inp)
        acc += y[(row * INP + inp) * Dm + d] * w[inp];
    codes[idx] = acc;
}

__global__ void k_add_codes(float* __restrict__ tok, const float* __restrict__ codes) {
    int idx = blockIdx.x * 256 + threadIdx.x;           // 128*64*256
    if (idx >= Bsz * NP * Dm) return;
    int d = idx & 255; int t = idx >> 8;
    int npp = t & 63; int b = t >> 6;
    int srcnp = (npp & 7) * 8 + (npp >> 3);             // grid transpose
    tok[idx] += codes[((long)srcnp * Bsz + b) * Dm + d];
}

__global__ void k_enc_lin(const float* __restrict__ y, const float* __restrict__ w,
                          const float* __restrict__ lb, float* __restrict__ coded) {
    int idx = blockIdx.x * 256 + threadIdx.x;           // 128*256
    if (idx >= Bsz * Dm) return;
    int d = idx & 255; int b = idx >> 8;
    float acc = lb[0];
    for (int np = 0; np < NP; ++np)
        acc += y[((long)b * NP + np) * Dm + d] * w[np];
    coded[idx] = acc;
}

__global__ void k_scatter_out(const float* __restrict__ up, const float* __restrict__ ub,
                              float* __restrict__ out) {
    int idx = blockIdx.x * 256 + threadIdx.x;           // 128*2*32*32
    if (idx >= Bsz * Cch * IMG * IMG) return;
    int q = idx & 31; int t = idx >> 5;
    int p = t & 31; int t2 = t >> 5;
    int c = t2 & 1; int b = t2 >> 1;
    int r = p >> 2, pp = p & 3, hc = q >> 2, qq = q & 3;
    int tokrow = b * NP + r * 8 + hc;
    int n = c * 16 + pp * 4 + qq;
    out[idx] = up[tokrow * 32 + n] + ub[c];
}

// ---------------------------------------------------------------------------
// Host-side helpers
// ---------------------------------------------------------------------------
static inline void launch_gemm(hipStream_t st, const GemmDesc& g) {
    dim3 grid((g.N + TN - 1) / TN, (g.M + TM - 1) / TM, g.G0 * g.G1 * g.G2);
    hipLaunchKernelGGL(k_gemm_bf16, grid, dim3(256), 0, st, g);
}

static void gemm2d(hipStream_t st, const float* A, long lda, const float* Bw, long bx,
                   const float* bias, float* Dst, long ldd,
                   int M, int N, int K, int relu, int transB, float alpha = 1.f) {
    GemmDesc g{};
    g.A = A; g.a0 = g.a1 = g.a2 = 0; g.am = lda;
    g.B = Bw; g.b0 = g.b1 = g.b2 = 0; g.bx = bx;
    g.bias = bias; g.D = Dst; g.d0 = g.d1 = g.d2 = 0; g.dm = ldd;
    g.M = M; g.N = N; g.K = K; g.G0 = g.G1 = g.G2 = 1;
    g.alpha = alpha; g.relu = relu; g.transB = transB;
    launch_gemm(st, g);
}

// batched attention core: seq = 128 (batch axis), head_dim = 32
static void attention(hipStream_t st,
                      const float* Q, long q0, long q1, long q2, long qm,
                      const float* Kp, long kg0, long kg1, long kg2, long kseq,
                      const float* Vp, long vg0, long vg1, long vg2, long vseq,
                      float* Sc, float* O, long o0, long o1, long o2, long om,
                      int G0, int G1, int G2, int causal) {
    int G = G0 * G1 * G2;
    GemmDesc gs{};
    gs.A = Q;  gs.a0 = q0; gs.a1 = q1; gs.a2 = q2; gs.am = qm;
    gs.B = Kp; gs.b0 = kg0; gs.b1 = kg1; gs.b2 = kg2; gs.bx = kseq; gs.transB = 1;
    gs.bias = nullptr;
    gs.D = Sc; gs.d2 = 128L * 128; gs.d1 = gs.d2 * G2; gs.d0 = gs.d1 * G1; gs.dm = 128;
    gs.M = 128; gs.N = 128; gs.K = HD; gs.G0 = G0; gs.G1 = G1; gs.G2 = G2;
    gs.alpha = 0.17677669529663687f;   // 1/sqrt(32)
    gs.relu = 0;
    launch_gemm(st, gs);

    hipLaunchKernelGGL(k_softmax128, dim3(G * 128), dim3(128), 0, st, Sc, 128, causal);

    GemmDesc go{};
    go.A = Sc; go.a2 = 128L * 128; go.a1 = go.a2 * G2; go.a0 = go.a1 * G1; go.am = 128;
    go.B = Vp; go.b0 = vg0; go.b1 = vg1; go.b2 = vg2; go.bx = vseq; go.transB = 0;
    go.bias = nullptr;
    go.D = O; go.d0 = o0; go.d1 = o1; go.d2 = o2; go.dm = om;
    go.M = 128; go.N = HD; go.K = 128; go.G0 = G0; go.G1 = G1; go.G2 = G2;
    go.alpha = 1.f; go.relu = 0;
    launch_gemm(st, go);
}

static void self_attn_layer(hipStream_t st, float* y, int rows,
                            const float* qkv_w, const float* qkv_b,
                            const float* out_w, const float* out_b,
                            const float* ln_g, const float* ln_b,
                            int G0, int G1, int G2,
                            long q0, long q1, long q2, long qm,
                            long o0, long o1, long o2, long om,
                            float* t_qkv, float* t_s, float* t_o, float* t_t, int causal) {
    gemm2d(st, y, Dm, qkv_w, Dm, qkv_b, t_qkv, 3 * Dm, rows, 3 * Dm, Dm, 0, 1);
    attention(st,
              t_qkv,           q0, q1, q2, qm,
              t_qkv + Dm,      q0, q1, q2, qm,
              t_qkv + 2 * Dm,  q0, q1, q2, qm,
              t_s, t_o, o0, o1, o2, om, G0, G1, G2, causal);
    gemm2d(st, t_o, Dm, out_w, Dm, out_b, t_t, Dm, rows, Dm, Dm, 0, 1);
    hipLaunchKernelGGL(k_add_ln, dim3(rows), dim3(256), 0, st, y, t_t, ln_g, ln_b);
}

static void ffn_layer(hipStream_t st, float* y, int rows,
                      const float* w1, const float* b1,
                      const float* w2, const float* b2,
                      const float* ln_g, const float* ln_b, float* t_h, float* t_t) {
    gemm2d(st, y, Dm, w1, Dm, b1, t_h, Ff, rows, Ff, Dm, 1, 1);
    gemm2d(st, t_h, Ff, w2, Ff, b2, t_t, Dm, rows, Dm, Ff, 0, 1);
    hipLaunchKernelGGL(k_add_ln, dim3(rows), dim3(256), 0, st, y, t_t, ln_g, ln_b);
}

// ---------------------------------------------------------------------------
extern "C" void kernel_launch(void* const* d_in, const int* in_sizes, int n_in,
                              void* d_out, int out_size, void* d_ws, size_t ws_size,
                              hipStream_t stream) {
    (void)in_sizes; (void)n_in; (void)out_size; (void)ws_size;

    const float* x            = (const float*)d_in[0];
    const float* enc_patch_w  = (const float*)d_in[1];
    const float* enc_patch_b  = (const float*)d_in[2];
    const float* enc_pos      = (const float*)d_in[3];
    const float* enc_qkv_w    = (const float*)d_in[4];
    const float* enc_qkv_b    = (const float*)d_in[5];
    const float* enc_out_w    = (const float*)d_in[6];
    const float* enc_out_b    = (const float*)d_in[7];
    const float* enc_ln1_g    = (const float*)d_in[8];
    const float* enc_ln1_b    = (const float*)d_in[9];
    const float* enc_ff1_w    = (const float*)d_in[10];
    const float* enc_ff1_b    = (const float*)d_in[11];
    const float* enc_ff2_w    = (const float*)d_in[12];
    const float* enc_ff2_b    = (const float*)d_in[13];
    const float* enc_ln2_g    = (const float*)d_in[14];
    const float* enc_ln2_b    = (const float*)d_in[15];
    const float* inner_patch_w= (const float*)d_in[16];
    const float* inner_patch_b= (const float*)d_in[17];
    const float* inner_pos    = (const float*)d_in[18];
    const float* inner_qkv_w  = (const float*)d_in[19];
    const float* inner_qkv_b  = (const float*)d_in[20];
    const float* inner_out_w  = (const float*)d_in[21];
    const float* inner_out_b  = (const float*)d_in[22];
    const float* inner_ln1_g  = (const float*)d_in[23];
    const float* inner_ln1_b  = (const float*)d_in[24];
    const float* inner_ff1_w  = (const float*)d_in[25];
    const float* inner_ff1_b  = (const float*)d_in[26];
    const float* inner_ff2_w  = (const float*)d_in[27];
    const float* inner_ff2_b  = (const float*)d_in[28];
    const float* inner_ln2_g  = (const float*)d_in[29];
    const float* inner_ln2_b  = (const float*)d_in[30];
    const float* enc_lin_w    = (const float*)d_in[31];
    const float* enc_lin_b    = (const float*)d_in[32];
    const float* inner_lin_w  = (const float*)d_in[33];
    const float* inner_lin_b  = (const float*)d_in[34];
    const float* dec_lin_w    = (const float*)d_in[35];
    const float* dec_lin_b    = (const float*)d_in[36];
    const float* dec_pos      = (const float*)d_in[37];
    const float* dec_sa_qkv_w = (const float*)d_in[38];
    const float* dec_sa_qkv_b = (const float*)d_in[39];
    const float* dec_sa_out_w = (const float*)d_in[40];
    const float* dec_sa_out_b = (const float*)d_in[41];
    const float* dec_ca_qkv_w = (const float*)d_in[42];
    const float* dec_ca_qkv_b = (const float*)d_in[43];
    const float* dec_ca_out_w = (const float*)d_in[44];
    const float* dec_ca_out_b = (const float*)d_in[45];
    const float* dec_ln1_g    = (const float*)d_in[46];
    const float* dec_ln1_b    = (const float*)d_in[47];
    const float* dec_ln2_g    = (const float*)d_in[48];
    const float* dec_ln2_b    = (const float*)d_in[49];
    const float* dec_ln3_g    = (const float*)d_in[50];
    const float* dec_ln3_b    = (const float*)d_in[51];
    const float* dec_ff1_w    = (const float*)d_in[52];
    const float* dec_ff1_b    = (const float*)d_in[53];
    const float* dec_ff2_w    = (const float*)d_in[54];
    const float* dec_ff2_b    = (const float*)d_in[55];
    const float* unpatch_w    = (const float*)d_in[56];
    const float* unpatch_b    = (const float*)d_in[57];

    // ---- workspace carve (floats) ----
    float* ws = (float*)d_ws;
    size_t cur = 0;
    auto alloc = [&](size_t n) { float* p = ws + cur; cur += n; return p; };
    float* t_im    = alloc(262144);                    // im2col (8192*32 == 32768*8)
    float* t_tok   = alloc((size_t)Bsz * NP * Dm);     // 2M
    float* t_y     = alloc((size_t)32768 * Dm);        // 8.4M (inner & enc activations)
    float* t_qkv   = alloc((size_t)32768 * 3 * Dm);    // 25M
    float* t_s     = alloc((size_t)2048 * 128 * 128);  // 33.5M (scores, max G)
    float* t_o     = alloc((size_t)32768 * Dm);        // 8.4M
    float* t_t     = alloc((size_t)32768 * Dm);        // 8.4M
    float* t_h     = alloc((size_t)32768 * Ff);        // 67M (FFN hidden; also CA kv)
    float* t_codes = alloc((size_t)NP * Bsz * Dm);     // 2M
    float* t_coded = alloc((size_t)Bsz * Dm);          // 32K
    float* t_mem   = alloc((size_t)Bsz * NP * Dm);     // 2M
    float* t_up    = alloc((size_t)Bsz * NP * 32);     // 256K

    const int rowsO = Bsz * NP;          // 8192 outer tokens
    const int rowsI = NP * Bsz * INP;    // 32768 inner tokens

    // ---- outer patch embed: tok = im2col(x) @ Wp^T + b + pos ----
    hipLaunchKernelGGL(k_im2col_outer, dim3(1024), dim3(256), 0, stream, x, t_im);
    gemm2d(stream, t_im, 32, enc_patch_w, 32, enc_patch_b, t_tok, Dm, rowsO, Dm, 32, 0, 1);
    hipLaunchKernelGGL(k_add_pos, dim3(rowsO), dim3(256), 0, stream, t_tok, enc_pos, rowsO * Dm, NP, 1.f);

    // ---- inner patch embed ----
    hipLaunchKernelGGL(k_im2col_inner, dim3(1024), dim3(256), 0, stream, x, t_im);
    gemm2d(stream, t_im, 8, inner_patch_w, 8, inner_patch_b, t_y, Dm, rowsI, Dm, 8, 0, 1);
    hipLaunchKernelGGL(k_add_pos, dim3(rowsI), dim3(256), 0, stream, t_y, inner_pos, rowsI * Dm, INP, 1.f);

    // ---- inner transformer: groups (np, inp, h), seq over batch ----
    for (int i = 0; i < Ll; ++i) {
        self_attn_layer(stream, t_y, rowsI,
                        inner_qkv_w + (size_t)i * 3 * Dm * Dm, inner_qkv_b + (size_t)i * 3 * Dm,
                        inner_out_w + (size_t)i * Dm * Dm,     inner_out_b + (size_t)i * Dm,
                        inner_ln1_g + (size_t)i * Dm,          inner_ln1_b + (size_t)i * Dm,
                        NP, INP, Hh,
                        /*qkv strides*/ (long)Bsz * INP * 768, 768, 32, (long)INP * 768,
                        /*o strides  */ (long)Bsz * INP * 256, 256, 32, (long)INP * 256,
                        t_qkv, t_s, t_o, t_t, 0);
        ffn_layer(stream, t_y, rowsI,
                  inner_ff1_w + (size_t)i * Ff * Dm, inner_ff1_b + (size_t)i * Ff,
                  inner_ff2_w + (size_t)i * Ff * Dm, inner_ff2_b + (size_t)i * Dm,
                  inner_ln2_g + (size_t)i * Dm,      inner_ln2_b + (size_t)i * Dm, t_h, t_t);
    }

    // ---- codes + grid-transposed add into tok ----
    hipLaunchKernelGGL(k_inner_lin, dim3(8192), dim3(256), 0, stream, t_y, inner_lin_w, inner_lin_b, t_codes);
    hipLaunchKernelGGL(k_add_codes, dim3(8192), dim3(256), 0, stream, t_tok, t_codes);

    // ---- outer encoder on a copy of tok ----
    hipMemcpyAsync(t_y, t_tok, (size_t)rowsO * Dm * sizeof(float), hipMemcpyDeviceToDevice, stream);
    for (int i = 0; i < Ll; ++i) {
        self_attn_layer(stream, t_y, rowsO,
                        enc_qkv_w + (size_t)i * 3 * Dm * Dm, enc_qkv_b + (size_t)i * 3 * Dm,
                        enc_out_w + (size_t)i * Dm * Dm,     enc_out_b + (size_t)i * Dm,
                        enc_ln1_g + (size_t)i * Dm,          enc_ln1_b + (size_t)i * Dm,
                        NP, 1, Hh,
                        768, 0, 32, (long)NP * 768,
                        256, 0, 32, (long)NP * 256,
                        t_qkv, t_s, t_o, t_t, 0);
        ffn_layer(stream, t_y, rowsO,
                  enc_ff1_w + (size_t)i * Ff * Dm, enc_ff1_b + (size_t)i * Ff,
                  enc_ff2_w + (size_t)i * Ff * Dm, enc_ff2_b + (size_t)i * Dm,
                  enc_ln2_g + (size_t)i * Dm,      enc_ln2_b + (size_t)i * Dm, t_h, t_t);
    }
    hipLaunchKernelGGL(k_enc_lin, dim3(128), dim3(256), 0, stream, t_y, enc_lin_w, enc_lin_b, t_coded);

    // ---- decoder memory: mem = coded @ dec_lin_w^T + b  -> [B, NP, D] ----
    gemm2d(stream, t_coded, Dm, dec_lin_w, Dm, dec_lin_b, t_mem, NP * Dm, Bsz, NP * Dm, Dm, 0, 1);

    // ---- decoder (in place on t_tok == nn_input) ----
    for (int i = 0; i < Ll; ++i) {
        // causal self-attention over batch axis
        self_attn_layer(stream, t_tok, rowsO,
                        dec_sa_qkv_w + (size_t)i * 3 * Dm * Dm, dec_sa_qkv_b + (size_t)i * 3 * Dm,
                        dec_sa_out_w + (size_t)i * Dm * Dm,     dec_sa_out_b + (size_t)i * Dm,
                        dec_ln1_g + (size_t)i * Dm,             dec_ln1_b + (size_t)i * Dm,
                        NP, 1, Hh,
                        768, 0, 32, (long)NP * 768,
                        256, 0, 32, (long)NP * 256,
                        t_qkv, t_s, t_o, t_t, 1);
        // cross-attention: q from y, k/v from mem
        gemm2d(stream, t_tok, Dm, dec_ca_qkv_w + (size_t)i * 3 * Dm * Dm, Dm,
               dec_ca_qkv_b + (size_t)i * 3 * Dm, t_qkv, 3 * Dm, rowsO, 3 * Dm, Dm, 0, 1);
        float* t_kv = t_h;   // [8192, 512] : k then v
        gemm2d(stream, t_mem, Dm, dec_ca_qkv_w + (size_t)i * 3 * Dm * Dm + (size_t)Dm * Dm, Dm,
               dec_ca_qkv_b + (size_t)i * 3 * Dm + Dm, t_kv, 2 * Dm, rowsO, 2 * Dm, Dm, 0, 1);
        attention(stream,
                  t_qkv,      768, 0, 32, (long)NP * 768,
                  t_kv,       512, 0, 32, (long)NP * 512,
                  t_kv + Dm,  512, 0, 32, (long)NP * 512,
                  t_s, t_o, 256, 0, 32, (long)NP * 256, NP, 1, Hh, 0);
        gemm2d(stream, t_o, Dm, dec_ca_out_w + (size_t)i * Dm * Dm, Dm,
               dec_ca_out_b + (size_t)i * Dm, t_t, Dm, rowsO, Dm, Dm, 0, 1);
        hipLaunchKernelGGL(k_add_ln, dim3(rowsO), dim3(256), 0, stream,
                           t_tok, t_t, dec_ln2_g + (size_t)i * Dm, dec_ln2_b + (size_t)i * Dm);
        // FFN
        ffn_layer(stream, t_tok, rowsO,
                  dec_ff1_w + (size_t)i * Ff * Dm, dec_ff1_b + (size_t)i * Ff,
                  dec_ff2_w + (size_t)i * Ff * Dm, dec_ff2_b + (size_t)i * Dm,
                  dec_ln3_g + (size_t)i * Dm,      dec_ln3_b + (size_t)i * Dm, t_h, t_t);
    }

    // ---- y - dec_pos, unpatch GEMM, scatter to image ----
    hipLaunchKernelGGL(k_add_pos, dim3(rowsO), dim3(256), 0, stream, t_tok, dec_pos, rowsO * Dm, NP, -1.f);
    gemm2d(stream, t_tok, Dm, unpatch_w, 32, nullptr, t_up, 32, rowsO, 32, Dm, 0, 0);
    hipLaunchKernelGGL(k_scatter_out, dim3(1024), dim3(256), 0, stream, t_up, unpatch_b, (float*)d_out);
}
